// GRU_88493506167244
// MI455X (gfx1250) — matrix operand order repaired and checked by reading, not compiled, not run on hardware
//
#include <hip/hip_runtime.h>
#include <hip/hip_bf16.h>
#include <math.h>

// ---------------------------------------------------------------------------
// GRU LM on MI455X (gfx1250, wave32, WMMA + async-to-LDS DMA).
//  - Precompute non-recurrent gate inputs Xall = emb @ We.T + bias  (WMMA bf16)
//  - Persistent 32-WG scan kernel with grid barriers: per step
//      rz = sigmoid(h @ Wrz_h.T + X_rz) ; hbar = tanh((r*h) @ Wbar_h.T + X_bar)
//      h  = (1-z)*h + z*hbar
//  - Head GEMM hidden @ Whead.T + bhead (WMMA bf16, fp32 out)
// ---------------------------------------------------------------------------

typedef __attribute__((ext_vector_type(16))) __bf16 v16bf;
typedef __attribute__((ext_vector_type(8)))  __bf16 v8bf;
typedef __attribute__((ext_vector_type(8)))  float  v8f;
typedef __attribute__((ext_vector_type(4)))  int    v4i;

constexpr int B_  = 64;
constexpr int T_  = 256;
constexpr int V_  = 4096;
constexpr int E_  = 512;
constexpr int H_  = 1024;
constexpr int BT_ = B_ * T_;         // 16384
constexpr int NG3 = 3 * H_;          // 3072 combined gate cols: [r | z | bar]
constexpr int NRZ = 2 * H_;          // 2048 combined r,z cols

// ---- workspace layout (all sizes multiples of 256 bytes) -------------------
constexpr size_t OFF_BAR  = 0;                                    // 256 B barrier
constexpr size_t OFF_H32  = 256;                                  // h fp32 [64][1024]
constexpr size_t OFF_HBF  = OFF_H32  + (size_t)B_ * H_ * 4;       // h bf16
constexpr size_t OFF_Z    = OFF_HBF  + (size_t)B_ * H_ * 2;       // z fp32
constexpr size_t OFF_RH   = OFF_Z    + (size_t)B_ * H_ * 4;       // r*h bf16
constexpr size_t OFF_BIAS = OFF_RH   + (size_t)B_ * H_ * 2;       // bias_all[3072] f32
constexpr size_t OFF_WE   = OFF_BIAS + (size_t)NG3 * 4;           // We_all [3072][512] bf16
constexpr size_t OFF_WRZ  = OFF_WE   + (size_t)NG3 * E_ * 2;      // Wrz_h [2048][1024] bf16
constexpr size_t OFF_WBAR = OFF_WRZ  + (size_t)NRZ * H_ * 2;      // Wbar_h [1024][1024] bf16
constexpr size_t OFF_WHD  = OFF_WBAR + (size_t)H_ * H_ * 2;       // Whead [4096][1024] bf16
constexpr size_t OFF_EMB  = OFF_WHD  + (size_t)V_ * H_ * 2;       // emb bf16 [16384][512] (time-major rows m=t*64+b)
constexpr size_t OFF_X    = OFF_EMB  + (size_t)BT_ * E_ * 2;      // Xall bf16 [16384][3072]
constexpr size_t OFF_HID  = OFF_X    + (size_t)BT_ * NG3 * 2;     // hidden bf16 [16384][1024]
constexpr size_t WS_NEED  = OFF_HID  + (size_t)BT_ * H_ * 2;      // ~170 MB

// ---- async global->LDS copy (gfx1250 GLOBAL_LOAD_ASYNC_TO_LDS_B128) --------
#if defined(__HIP_DEVICE_COMPILE__) && __has_builtin(__builtin_amdgcn_global_load_async_to_lds_b128)
#define ASYNC_LDS 1
#else
#define ASYNC_LDS 0
#endif

typedef __attribute__((address_space(1))) v4i as1_v4i;  // global 16B vector
typedef __attribute__((address_space(3))) v4i as3_v4i;  // LDS 16B vector

__device__ __forceinline__ void cp16_g2l(const __bf16* g, __bf16* l) {
#if ASYNC_LDS
  __builtin_amdgcn_global_load_async_to_lds_b128((as1_v4i*)g, (as3_v4i*)l, 0, 0);
#else
  *(v8bf*)l = *(const v8bf*)g;
#endif
}

template <int N>
__device__ __forceinline__ void cp_wait() {
#if ASYNC_LDS
#if __has_builtin(__builtin_amdgcn_s_wait_asynccnt)
  __builtin_amdgcn_s_wait_asynccnt(N);
#else
  asm volatile("s_wait_asynccnt %0" ::"i"(N) : "memory");
#endif
#endif
}

// ---- WMMA helpers ----------------------------------------------------------
__device__ __forceinline__ v8f wmma_bf16(v16bf a, v16bf b, v8f c) {
  // emits v_wmma_f32_16x16x32_bf16
  return __builtin_amdgcn_wmma_f32_16x16x32_bf16(false, a, false, b, (short)0, c,
                                                 false, false);
}
__device__ __forceinline__ v16bf cat8(v8bf lo, v8bf hi) {
  return __builtin_shufflevector(lo, hi, 0,1,2,3,4,5,6,7,8,9,10,11,12,13,14,15);
}

// ---- grid barrier for the persistent scan kernel ---------------------------
struct GridBar { unsigned cnt; unsigned gen; };

__device__ __forceinline__ void grid_barrier(GridBar* gb, unsigned nwg) {
  __syncthreads();
  if (threadIdx.x == 0) {
    __threadfence();  // make this WG's global stores visible device-wide
    unsigned g = __hip_atomic_load(&gb->gen, __ATOMIC_RELAXED, __HIP_MEMORY_SCOPE_AGENT);
    unsigned a = __hip_atomic_fetch_add(&gb->cnt, 1u, __ATOMIC_ACQ_REL, __HIP_MEMORY_SCOPE_AGENT);
    if (a == nwg - 1u) {
      __hip_atomic_store(&gb->cnt, 0u, __ATOMIC_RELAXED, __HIP_MEMORY_SCOPE_AGENT);
      __hip_atomic_fetch_add(&gb->gen, 1u, __ATOMIC_RELEASE, __HIP_MEMORY_SCOPE_AGENT);
    } else {
      while (__hip_atomic_load(&gb->gen, __ATOMIC_ACQUIRE, __HIP_MEMORY_SCOPE_AGENT) == g) {
        __builtin_amdgcn_s_sleep(2);
      }
    }
  }
  __syncthreads();
}

// ===========================================================================
// Prep kernels: fp32 -> bf16 weight repacks, embedding gather, h init
// ===========================================================================
__global__ void k_prep_we(const float* __restrict__ Wr, const float* __restrict__ Wz,
                          const float* __restrict__ Wbar,
                          const float* __restrict__ br, const float* __restrict__ bz,
                          const float* __restrict__ bbar,
                          __bf16* __restrict__ We_all, float* __restrict__ bias_all) {
  int idx = blockIdx.x * blockDim.x + threadIdx.x;
  if (idx < NG3 * E_) {
    int n = idx >> 9, k = idx & (E_ - 1);
    int g = n >> 10, u = n & (H_ - 1);
    const float* W = (g == 0) ? Wr : ((g == 1) ? Wz : Wbar);
    We_all[idx] = (__bf16)W[(size_t)u * (E_ + H_) + k];
  }
  if (idx < NG3) {
    int g = idx >> 10, u = idx & (H_ - 1);
    bias_all[idx] = (g == 0) ? br[u] : ((g == 1) ? bz[u] : bbar[u]);
  }
}

__global__ void k_prep_wrz(const float* __restrict__ Wr, const float* __restrict__ Wz,
                           __bf16* __restrict__ Wrz) {
  int idx = blockIdx.x * blockDim.x + threadIdx.x;
  if (idx >= NRZ * H_) return;
  int n = idx >> 10, k = idx & (H_ - 1);
  const float* W = (n < H_) ? (Wr + (size_t)n * (E_ + H_))
                            : (Wz + (size_t)(n - H_) * (E_ + H_));
  Wrz[idx] = (__bf16)W[E_ + k];
}

__global__ void k_prep_wbar(const float* __restrict__ Wbar, __bf16* __restrict__ Wb) {
  int idx = blockIdx.x * blockDim.x + threadIdx.x;
  if (idx >= H_ * H_) return;
  int n = idx >> 10, k = idx & (H_ - 1);
  Wb[idx] = (__bf16)Wbar[(size_t)n * (E_ + H_) + E_ + k];
}

__global__ void k_prep_whead(const float* __restrict__ Wh, __bf16* __restrict__ o) {
  int idx = blockIdx.x * blockDim.x + threadIdx.x;
  if (idx >= V_ * H_) return;
  o[idx] = (__bf16)Wh[idx];
}

__global__ void k_embed(const int* __restrict__ x, const float* __restrict__ wte,
                        __bf16* __restrict__ emb) {
  int idx = blockIdx.x * blockDim.x + threadIdx.x;
  if (idx >= BT_ * E_) return;
  int m = idx >> 9, e = idx & (E_ - 1);
  int b = m & (B_ - 1), t = m >> 6;            // time-major rows: m = t*64 + b
  int tok = x[b * T_ + t];
  emb[idx] = (__bf16)wte[(size_t)tok * E_ + e];
}

__global__ void k_init_h(const float* __restrict__ start, float* __restrict__ h32,
                         __bf16* __restrict__ hbf) {
  int idx = blockIdx.x * blockDim.x + threadIdx.x;
  if (idx >= B_ * H_) return;
  float v = start[idx & (H_ - 1)];
  h32[idx] = v;
  hbf[idx] = (__bf16)v;
}

// ===========================================================================
// Generic tiled WMMA GEMM:  out[m][n] = A[m][:] . Bm[n][:] + bias[n]
//   A: [M][K] bf16 row-major; Bm: [N][K] bf16 (i.e. W in its native layout)
//   WG tile 128(M) x 64(N), 8 waves of 32x32, K step 32.
//   A staged via async global->LDS DMA, double-buffered so the DMA for tile
//   k+1 overlaps the 4 WMMAs of tile k (ASYNCcnt completes in order).
// ===========================================================================
template <bool OUT_BF16, bool REMAP>
__global__ __launch_bounds__(256) void gemm_bias(const __bf16* __restrict__ A,
                                                 const __bf16* __restrict__ Bm,
                                                 const float* __restrict__ bias,
                                                 void* __restrict__ out,
                                                 int K, int N) {
  __shared__ __bf16 As[2][128 * 40];  // 40-half padded stride -> conflict-free
  const int tid   = threadIdx.x;
  const int wave  = tid >> 5, lane = tid & 31;
  const int laneN = lane & 15, half = lane >> 4;
  const int mbase = blockIdx.y * 128;
  const int nbase = blockIdx.x * 64;
  const int m0    = 32 * (wave & 3);
  const int n0    = 32 * (wave >> 2);

  auto stage = [&](int k0, __bf16* dst) {
    // 128x32 bf16 tile, 2 async b128 ops per thread
    #pragma unroll
    for (int c = tid; c < 512; c += 256) {
      int row = c >> 2, part = c & 3;
      cp16_g2l(A + (size_t)(mbase + row) * K + k0 + part * 8,
               dst + row * 40 + part * 8);
    }
  };

  v8f z8 = {0,0,0,0,0,0,0,0};
  v8f acc00 = z8, acc01 = z8, acc10 = z8, acc11 = z8;

  stage(0, As[0]);
  const int nk = K >> 5;
  for (int ki = 0; ki < nk; ++ki) {
    const int k0 = ki << 5;
    __bf16* cur = As[ki & 1];
    if (ki + 1 < nk) {
      stage(k0 + 32, As[(ki + 1) & 1]);
      cp_wait<2>();          // wait for current tile; next tile stays in flight
    } else {
      cp_wait<0>();
    }
    __syncthreads();

    const __bf16* bp = Bm + (size_t)(nbase + n0 + laneN) * K + k0 + half * 16;
    if (k0 + 32 < K) __builtin_prefetch(bp + 32, 0, 1);   // global_prefetch_b8
    v16bf b0 = *(const v16bf*)bp;
    v16bf b1 = *(const v16bf*)(bp + (size_t)16 * K);

    const __bf16* a0p = cur + (m0 + laneN) * 40 + half * 16;
    v16bf a0 = cat8(*(const v8bf*)a0p, *(const v8bf*)(a0p + 8));
    const __bf16* a1p = cur + (m0 + 16 + laneN) * 40 + half * 16;
    v16bf a1 = cat8(*(const v8bf*)a1p, *(const v8bf*)(a1p + 8));

    acc00 = wmma_bf16(a0, b0, acc00);
    acc01 = wmma_bf16(a0, b1, acc01);
    acc10 = wmma_bf16(a1, b0, acc10);
    acc11 = wmma_bf16(a1, b1, acc11);
    __syncthreads();         // all waves done reading 'cur' before it's re-staged
  }

  // epilogue
  #pragma unroll
  for (int mt = 0; mt < 2; ++mt) {
    #pragma unroll
    for (int nt = 0; nt < 2; ++nt) {
      v8f acc = (mt == 0) ? ((nt == 0) ? acc00 : acc01)
                          : ((nt == 0) ? acc10 : acc11);
      #pragma unroll
      for (int i = 0; i < 8; ++i) {
        int m = mbase + m0 + mt * 16 + i + 8 * half;
        int n = nbase + n0 + nt * 16 + laneN;
        float v = acc[i] + bias[n];
        size_t orow = REMAP ? ((size_t)(m & (B_ - 1)) * T_ + (m >> 6)) : (size_t)m;
        if constexpr (OUT_BF16) ((__bf16*)out)[orow * (size_t)N + n] = (__bf16)v;
        else                    ((float*)out)[orow * (size_t)N + n]  = v;
      }
    }
  }
}

// ===========================================================================
// Persistent GRU scan: 32 WGs x 256 threads, T=256 steps, 2 grid barriers/step
// ===========================================================================
__global__ __launch_bounds__(256, 1) void gru_scan(const __bf16* __restrict__ Xall,
                                                   const __bf16* __restrict__ Wrz,
                                                   const __bf16* __restrict__ Wbh,
                                                   float* __restrict__ h32,
                                                   __bf16* __restrict__ hbf,
                                                   float* __restrict__ zbuf,
                                                   __bf16* __restrict__ rhbuf,
                                                   __bf16* __restrict__ hidden,
                                                   GridBar* gb) {
  extern __shared__ char smem[];
  __bf16* hl = (__bf16*)smem;                      // [64][1032] padded bf16
  const int tid   = threadIdx.x;
  const int g     = blockIdx.x;                    // 0..31
  const int wave  = tid >> 5, lane = tid & 31;
  const int laneN = lane & 15, half = lane >> 4;
  v8f z8 = {0,0,0,0,0,0,0,0};

  for (int t = 0; t < T_; ++t) {
    // ---- Phase A: async-stage h (bf16) into LDS ---------------------------
    for (int c = tid; c < 64 * 128; c += 256) {
      int row = c >> 7, part = c & 127;
      cp16_g2l(hbf + row * H_ + part * 8, hl + row * 1032 + part * 8);
    }
    cp_wait<0>();
    __syncthreads();

    // ---- Phase B: rz GEMM, WG covers combined cols [64g, 64g+64) ----------
    {
      const int n0 = g * 64 + 16 * (wave & 3);     // combined col base (r: <1024, z: >=1024)
      const int m0 = 32 * (wave >> 2);             // batch-row base
      v8f acc0 = z8, acc1 = z8;
      for (int k0 = 0; k0 < H_; k0 += 32) {
        const __bf16* bp = Wrz + (size_t)(n0 + laneN) * H_ + k0 + half * 16;
        if (k0 + 32 < H_) __builtin_prefetch(bp + 32, 0, 1);
        v16bf bf = *(const v16bf*)bp;
        const __bf16* a0p = hl + (m0 + laneN) * 1032 + k0 + half * 16;
        v16bf a0 = cat8(*(const v8bf*)a0p, *(const v8bf*)(a0p + 8));
        const __bf16* a1p = hl + (m0 + 16 + laneN) * 1032 + k0 + half * 16;
        v16bf a1 = cat8(*(const v8bf*)a1p, *(const v8bf*)(a1p + 8));
        acc0 = wmma_bf16(a0, bf, acc0);
        acc1 = wmma_bf16(a1, bf, acc1);
      }
      #pragma unroll
      for (int i = 0; i < 8; ++i) {
        #pragma unroll
        for (int mt = 0; mt < 2; ++mt) {
          float av = (mt == 0) ? acc0[i] : acc1[i];
          int m  = m0 + mt * 16 + i + 8 * half;
          int nc = n0 + laneN;
          float xv = (float)Xall[(size_t)(t * B_ + m) * NG3 + nc];
          float s  = 1.0f / (1.0f + __expf(-(av + xv)));
          if (nc < H_) {                           // r gate -> store r*h
            float ho = h32[m * H_ + nc];
            rhbuf[m * H_ + nc] = (__bf16)(s * ho);
          } else {                                 // z gate
            zbuf[m * H_ + (nc - H_)] = s;
          }
        }
      }
    }
    grid_barrier(gb, 32);

    // ---- Phase C: async-stage r*h, hbar GEMM, h update; cols [32g, 32g+32) -
    for (int c = tid; c < 64 * 128; c += 256) {
      int row = c >> 7, part = c & 127;
      cp16_g2l(rhbuf + row * H_ + part * 8, hl + row * 1032 + part * 8);
    }
    cp_wait<0>();
    __syncthreads();
    {
      const int m0 = 16 * (wave & 3);
      const int n0 = 32 * g + 16 * (wave >> 2);
      v8f acc = z8;
      for (int k0 = 0; k0 < H_; k0 += 32) {
        const __bf16* bp = Wbh + (size_t)(n0 + laneN) * H_ + k0 + half * 16;
        if (k0 + 32 < H_) __builtin_prefetch(bp + 32, 0, 1);
        v16bf bf = *(const v16bf*)bp;
        const __bf16* ap = hl + (m0 + laneN) * 1032 + k0 + half * 16;
        v16bf af = cat8(*(const v8bf*)ap, *(const v8bf*)(ap + 8));
        acc = wmma_bf16(af, bf, acc);
      }
      #pragma unroll
      for (int i = 0; i < 8; ++i) {
        int m = m0 + i + 8 * half;
        int n = n0 + laneN;
        float xv = (float)Xall[(size_t)(t * B_ + m) * NG3 + 2 * H_ + n];
        float hb = tanhf(acc[i] + xv);
        float zz = zbuf[m * H_ + n];
        float ho = h32[m * H_ + n];
        float hn = (1.0f - zz) * ho + zz * hb;
        h32[m * H_ + n] = hn;
        hbf[m * H_ + n] = (__bf16)hn;
        hidden[(size_t)(t * B_ + m) * H_ + n] = (__bf16)hn;
      }
    }
    grid_barrier(gb, 32);
  }
}

// ===========================================================================
extern "C" void kernel_launch(void* const* d_in, const int* in_sizes, int n_in,
                              void* d_out, int out_size, void* d_ws, size_t ws_size,
                              hipStream_t stream) {
  const int*   x     = (const int*)d_in[0];
  const float* start = (const float*)d_in[1];
  const float* wte   = (const float*)d_in[2];
  const float* Wr    = (const float*)d_in[3];
  const float* br    = (const float*)d_in[4];
  const float* Wbar  = (const float*)d_in[5];
  const float* bbar  = (const float*)d_in[6];
  const float* Wz    = (const float*)d_in[7];
  const float* bz    = (const float*)d_in[8];
  const float* Whead = (const float*)d_in[9];
  const float* bhead = (const float*)d_in[10];

  char* ws = (char*)d_ws;
  GridBar* gb       = (GridBar*)(ws + OFF_BAR);
  float*   h32      = (float*)  (ws + OFF_H32);
  __bf16*  hbf      = (__bf16*) (ws + OFF_HBF);
  float*   zbuf     = (float*)  (ws + OFF_Z);
  __bf16*  rhbuf    = (__bf16*) (ws + OFF_RH);
  float*   bias_all = (float*)  (ws + OFF_BIAS);
  __bf16*  We_all   = (__bf16*) (ws + OFF_WE);
  __bf16*  Wrz_h    = (__bf16*) (ws + OFF_WRZ);
  __bf16*  Wbar_h   = (__bf16*) (ws + OFF_WBAR);
  __bf16*  Whd_bf   = (__bf16*) (ws + OFF_WHD);
  __bf16*  emb      = (__bf16*) (ws + OFF_EMB);
  __bf16*  Xall     = (__bf16*) (ws + OFF_X);
  __bf16*  hidden   = (__bf16*) (ws + OFF_HID);
  (void)in_sizes; (void)n_in; (void)out_size; (void)ws_size; (void)WS_NEED;

  (void)hipMemsetAsync(ws + OFF_BAR, 0, 256, stream);

  const int TPB = 256;
  k_prep_we  <<<(NG3 * E_ + TPB - 1) / TPB, TPB, 0, stream>>>(Wr, Wz, Wbar, br, bz, bbar, We_all, bias_all);
  k_prep_wrz <<<(NRZ * H_ + TPB - 1) / TPB, TPB, 0, stream>>>(Wr, Wz, Wrz_h);
  k_prep_wbar<<<(H_ * H_  + TPB - 1) / TPB, TPB, 0, stream>>>(Wbar, Wbar_h);
  k_prep_whead<<<(V_ * H_ + TPB - 1) / TPB, TPB, 0, stream>>>(Whead, Whd_bf);
  k_embed    <<<(BT_ * E_ + TPB - 1) / TPB, TPB, 0, stream>>>(x, wte, emb);
  k_init_h   <<<(B_ * H_  + TPB - 1) / TPB, TPB, 0, stream>>>(start, h32, hbf);

  // Xall = emb @ We_all.T + bias_all   (M=16384, K=512, N=3072) -> bf16
  gemm_bias<true, false><<<dim3(NG3 / 64, BT_ / 128), TPB, 0, stream>>>(
      emb, We_all, bias_all, (void*)Xall, E_, NG3);

  // persistent GRU scan (32 WGs, dyn LDS = 64*1032*2 bytes)
  gru_scan<<<32, TPB, 64 * 1032 * 2, stream>>>(Xall, Wrz_h, Wbar_h, h32, hbf,
                                               zbuf, rhbuf, hidden, gb);

  // logits = hidden @ Whead.T + bhead  (M=16384, K=1024, N=4096) -> fp32, row remap
  gemm_bias<false, true><<<dim3(V_ / 64, BT_ / 128), TPB, 0, stream>>>(
      hidden, Whd_bf, bhead, d_out, H_, V_);
}
